// DGCNN_semseg_79585743994956
// MI455X (gfx1250) — matrix operand order repaired and checked by reading, not compiled, not run on hardware
//
#include <hip/hip_runtime.h>
#include <hip/hip_bf16.h>

typedef __attribute__((ext_vector_type(16))) _Float16 v16h;
typedef __attribute__((ext_vector_type(8)))  _Float16 v8h;
typedef __attribute__((ext_vector_type(8)))  float    v8f;

#define BATCH 4
#define NPTS  4096
#define KNN   20
#define NKCOL (NPTS * KNN)   // 81920 columns for edge-conv GEMMs
#define WPITCH 40            // LDS row pitch in halves: 80B, conflict-free b128 reads

// ---------------------------------------------------------------------------
// KNN: register top-20 by -||pi-pj||^2, gated unrolled compare-swap insertion.
// ---------------------------------------------------------------------------
__global__ void __launch_bounds__(256)
knn_kernel(const float* __restrict__ src, int* __restrict__ idxout,
           int batchStride, int cStride, int nStride)
{
    __shared__ float tile[3][256];
    const int b = blockIdx.y;
    const int n = blockIdx.x * 256 + threadIdx.x;
    const float* base = src + (size_t)b * batchStride;

    const float px = base[(size_t)n * nStride + 0 * cStride];
    const float py = base[(size_t)n * nStride + 1 * cStride];
    const float pz = base[(size_t)n * nStride + 2 * cStride];

    float vals[KNN];
    int   idxs[KNN];
#pragma unroll
    for (int s = 0; s < KNN; ++s) { vals[s] = -3.4e38f; idxs[s] = 0; }

    for (int j0 = 0; j0 < NPTS; j0 += 256) {
        const int jt = j0 + threadIdx.x;
        tile[0][threadIdx.x] = base[(size_t)jt * nStride + 0 * cStride];
        tile[1][threadIdx.x] = base[(size_t)jt * nStride + 1 * cStride];
        tile[2][threadIdx.x] = base[(size_t)jt * nStride + 2 * cStride];
        __syncthreads();
        for (int jj = 0; jj < 256; ++jj) {
            const float dx = px - tile[0][jj];
            const float dy = py - tile[1][jj];
            const float dz = pz - tile[2][jj];
            const float d  = -(dx * dx + dy * dy + dz * dz);
            if (d > vals[KNN - 1]) {
                float cv = d; int ci = j0 + jj;
#pragma unroll
                for (int s = 0; s < KNN; ++s) {
                    if (cv > vals[s]) {
                        float tv = vals[s]; int ti = idxs[s];
                        vals[s] = cv; idxs[s] = ci; cv = tv; ci = ti;
                    }
                }
            }
        }
        __syncthreads();
    }
    int* op = idxout + ((size_t)b * NPTS + n) * KNN;
#pragma unroll
    for (int s = 0; s < KNN; ++s) op[s] = idxs[s];
}

// ---------------------------------------------------------------------------
// Gather neighbor features: out[b,c,n*K+k] = src[b,c,idx[b,n,k]]
// ---------------------------------------------------------------------------
__global__ void gather_kernel(const float* __restrict__ src, const int* __restrict__ idx,
                              float* __restrict__ out, int C,
                              int batchStride, int cStride, int nStride, size_t total)
{
    size_t tid = (size_t)blockIdx.x * 256 + threadIdx.x;
    if (tid >= total) return;
    int k = (int)(tid % KNN); size_t r = tid / KNN;
    int n = (int)(r % NPTS);  r /= NPTS;
    int c = (int)(r % C);     int b = (int)(r / C);
    int j = idx[((size_t)b * NPTS + n) * KNN + k];
    out[tid] = src[(size_t)b * batchStride + (size_t)c * cStride + (size_t)j * nStride];
}

// ---------------------------------------------------------------------------
// Per-point bias: t[b,o,n] = sum_c (wc[o,c]-wd[o,c]) * x[b,c,n]
// ---------------------------------------------------------------------------
__global__ void bias_t_kernel(const float* __restrict__ W, int ldw, int C,
                              const float* __restrict__ x,
                              int batchStride, int cStride, int nStride,
                              float* __restrict__ t)
{
    int tid = blockIdx.x * 256 + threadIdx.x;   // B*64*N threads
    int n = tid % NPTS; int r = tid / NPTS;
    int o = r % 64;     int b = r / 64;
    const float* xb = x + (size_t)b * batchStride;
    float acc = 0.f;
    for (int c = 0; c < C; ++c)
        acc += (W[o * ldw + C + c] - W[o * ldw + c]) * xb[(size_t)c * cStride + (size_t)n * nStride];
    t[tid] = acc;
}

// ---------------------------------------------------------------------------
// Stage-1 edge conv (C=4, too small for WMMA): out[b,o,col] = wd.gx + t
// o is uniform per 256-thread block -> weight reads become scalar loads.
// ---------------------------------------------------------------------------
__global__ void edge1_kernel(const float* __restrict__ w1,
                             const float* __restrict__ gx,   // (B,4,NKCOL)
                             const float* __restrict__ t,    // (B,64,N)
                             float* __restrict__ out)        // (B,64,NKCOL)
{
    size_t tid = (size_t)blockIdx.x * 256 + threadIdx.x;     // B*64*NKCOL
    int col = (int)(tid % NKCOL); size_t r = tid / NKCOL;
    int o = (int)(r % 64); int b = (int)(r / 64);
    const float* g = gx + (size_t)b * 4 * NKCOL;
    float acc = w1[o * 8 + 0] * g[col]
              + w1[o * 8 + 1] * g[NKCOL + col]
              + w1[o * 8 + 2] * g[2 * NKCOL + col]
              + w1[o * 8 + 3] * g[3 * NKCOL + col];
    acc += t[((size_t)b * 64 + o) * NPTS + col / KNN];
    out[tid] = acc;
}

// ---------------------------------------------------------------------------
// v16h from two contiguous 16B LDS chunks (2x ds_load_b128 + shuffle)
// ---------------------------------------------------------------------------
__device__ __forceinline__ v16h ld2x8(const _Float16* p0, const _Float16* p1)
{
    v8h a = *(const v8h*)p0;
    v8h b = *(const v8h*)p1;
    return __builtin_shufflevector(a, b, 0, 1, 2, 3, 4, 5, 6, 7,
                                         8, 9, 10, 11, 12, 13, 14, 15);
}

// ---------------------------------------------------------------------------
// WMMA GEMM:  Y[b,o,m] = sum_c W[o,c] * X[b,c,m]  (+ per-point bias if KDIV>0)
// REQUIRES C % 32 == 0. Block 256 thr (8 waves), tile 64(O) x 64(M); each wave
// owns one 16x16 O-subtile x two M-subtiles -> 2 v_wmma per K-step.
// Software-pipelined: tile i+1 global loads are issued before tile i's WMMAs,
// hiding HBM/L2 latency behind the matrix ops (single LDS buffer).
// KDIV is compile time -> bias column division is a mul-shift, no v_rcp chain.
// ---------------------------------------------------------------------------
template <int KDIV>
__global__ void __launch_bounds__(256)
gemm_wmma_kernel(const float* __restrict__ W, int ldw, int O, int C,
                 const float* __restrict__ X, int M,
                 float* __restrict__ Y,
                 const float* __restrict__ bias)
{
    __shared__ _Float16 Wt [64][WPITCH];   // [o][k]
    __shared__ _Float16 XtT[64][WPITCH];   // [n][k]  (transposed tile)

    const int b    = blockIdx.z;
    const int n0   = blockIdx.x * 64;
    const int o0   = blockIdx.y * 64;
    const int tid  = threadIdx.x;
    const int lane = tid & 31;
    const int wave = tid >> 5;
    const int wo   = wave & 3;    // O sub-tile 0..3
    const int wq   = wave >> 2;   // M half 0..1 (two 16-wide subtiles each)

    const float* Xb = X + (size_t)b * C * M;
    v8f acc0 = {}, acc1 = {};

    const int wr  = tid >> 2;          // W stage: row 0..63
    const int wc8 = (tid & 3) * 8;     //          k-octet
    const int xn  = tid & 63;          // X stage: column n 0..63
    const int xc0 = (tid >> 6) * 8;    //          k-octet 0/8/16/24

    // fragment addresses (loop-invariant)
    const int m     = wo * 16 + (lane & 15);
    const int kb    = (lane < 16) ? 0 : 8;
    const int kOffB = (lane < 16) ? 0 : 16;
    const int nA    = 32 * wq + (lane & 15);

    // ---- prologue: load tile kk=0 into registers ----
    float4 wv0, wv1;
    float  xv[8];
    {
        const float* wp = W + (size_t)(o0 + wr) * ldw + wc8;
        wv0 = *(const float4*)wp;
        wv1 = *(const float4*)(wp + 4);
#pragma unroll
        for (int i = 0; i < 8; ++i)
            xv[i] = Xb[(size_t)(xc0 + i) * M + n0 + xn];
    }

    for (int kk = 0; kk < C; kk += 32) {
        // ---- commit registers -> LDS (converted to f16) ----
        {
            v8h h = { (_Float16)wv0.x, (_Float16)wv0.y, (_Float16)wv0.z, (_Float16)wv0.w,
                      (_Float16)wv1.x, (_Float16)wv1.y, (_Float16)wv1.z, (_Float16)wv1.w };
            *(v8h*)&Wt[wr][wc8] = h;
            v8h hx = { (_Float16)xv[0], (_Float16)xv[1], (_Float16)xv[2], (_Float16)xv[3],
                       (_Float16)xv[4], (_Float16)xv[5], (_Float16)xv[6], (_Float16)xv[7] };
            *(v8h*)&XtT[xn][xc0] = hx;
        }
        __syncthreads();

        // ---- prefetch tile kk+32 (clamped; issued before the WMMAs) ----
        {
            const int kn = min(kk + 32, C - 32);
            const float* wp = W + (size_t)(o0 + wr) * ldw + kn + wc8;
            wv0 = *(const float4*)wp;
            wv1 = *(const float4*)(wp + 4);
#pragma unroll
            for (int i = 0; i < 8; ++i)
                xv[i] = Xb[(size_t)(kn + xc0 + i) * M + n0 + xn];
        }

        // ---- fragments (contiguous 16B LDS reads) + 2x WMMA ----
        v16h a  = ld2x8(&Wt[m][kOffB], &Wt[m][kOffB + 8]);
        v16h b0 = ld2x8(&XtT[nA][kOffB],      &XtT[nA][kOffB + 8]);
        v16h b1 = ld2x8(&XtT[nA + 16][kOffB], &XtT[nA + 16][kOffB + 8]);

        acc0 = __builtin_amdgcn_wmma_f32_16x16x32_f16(false, a, false, b0, (short)0, acc0, false, false);
        acc1 = __builtin_amdgcn_wmma_f32_16x16x32_f16(false, a, false, b1, (short)0, acc1, false, false);
        __syncthreads();
    }

    // ---- epilogue: C/D layout — VGPR r, lane l: M = r + (l<16?0:8), N = l&15
    const int col0      = n0 + 32 * wq + (lane & 15);
    const int orow_base = o0 + wo * 16 + ((lane < 16) ? 0 : 8);
    float* Yb = Y + (size_t)b * O * M;
#pragma unroll
    for (int r = 0; r < 8; ++r) {
        const int orow = orow_base + r;
        float v0 = acc0[r], v1 = acc1[r];
        if constexpr (KDIV > 0) {
            const float* bb = bias + ((size_t)b * 64 + orow) * NPTS;
            v0 += bb[col0 / KDIV];            // compile-time divisor -> mul/shift
            v1 += bb[(col0 + 16) / KDIV];
        }
        Yb[(size_t)orow * M + col0]      = v0;
        Yb[(size_t)orow * M + col0 + 16] = v1;
    }
}

// ---------------------------------------------------------------------------
// Instance norm (one (b,channel) plane of M elems per block) + LeakyReLU
// ---------------------------------------------------------------------------
__global__ void __launch_bounds__(256)
instnorm_lrelu_kernel(float* __restrict__ Y, int M)
{
    __shared__ float sbuf[2][256];
    __shared__ float smv[2];
    const size_t base = (size_t)blockIdx.x * M;
    float s = 0.f, ss = 0.f;
    for (int i = threadIdx.x; i < M; i += 256) {
        float v = Y[base + i];
        s += v; ss += v * v;
    }
    sbuf[0][threadIdx.x] = s; sbuf[1][threadIdx.x] = ss;
    __syncthreads();
    for (int st = 128; st > 0; st >>= 1) {
        if (threadIdx.x < st) {
            sbuf[0][threadIdx.x] += sbuf[0][threadIdx.x + st];
            sbuf[1][threadIdx.x] += sbuf[1][threadIdx.x + st];
        }
        __syncthreads();
    }
    if (threadIdx.x == 0) {
        float mean = sbuf[0][0] / (float)M;
        float var  = sbuf[1][0] / (float)M - mean * mean;
        smv[0] = mean;
        smv[1] = rsqrtf(var + 1e-5f);
    }
    __syncthreads();
    const float mean = smv[0], rs = smv[1];
    for (int i = threadIdx.x; i < M; i += 256) {
        float v = (Y[base + i] - mean) * rs;
        Y[base + i] = (v >= 0.f) ? v : 0.2f * v;
    }
}

// max over k: (b,c,n*K+k) -> (b,c,n)
__global__ void maxk_kernel(const float* __restrict__ in, float* __restrict__ out, size_t total)
{
    size_t tid = (size_t)blockIdx.x * 256 + threadIdx.x;
    if (tid >= total) return;
    const float* p = in + tid * KNN;
    float m = p[0];
#pragma unroll
    for (int k = 1; k < KNN; ++k) m = fmaxf(m, p[k]);
    out[tid] = m;
}

// max over points: one block per (b, channel)
__global__ void __launch_bounds__(256)
maxn_kernel(const float* __restrict__ in, float* __restrict__ out)
{
    __shared__ float sbuf[256];
    const size_t base = (size_t)blockIdx.x * NPTS;
    float m = -3.4e38f;
    for (int i = threadIdx.x; i < NPTS; i += 256) m = fmaxf(m, in[base + i]);
    sbuf[threadIdx.x] = m;
    __syncthreads();
    for (int st = 128; st > 0; st >>= 1) {
        if (threadIdx.x < st) sbuf[threadIdx.x] = fmaxf(sbuf[threadIdx.x], sbuf[threadIdx.x + st]);
        __syncthreads();
    }
    if (threadIdx.x == 0) out[blockIdx.x] = sbuf[0];
}

// cat192[b, c, n] = {x1,x2,x3}
__global__ void build_cat_kernel(const float* __restrict__ x1, const float* __restrict__ x2,
                                 const float* __restrict__ x3, float* __restrict__ out)
{
    size_t tid = (size_t)blockIdx.x * 256 + threadIdx.x;
    if (tid >= (size_t)BATCH * 192 * NPTS) return;
    int n = (int)(tid % NPTS); size_t r = tid / NPTS;
    int c = (int)(r % 192);    int b = (int)(r / 192);
    const float* src; int cc;
    if (c < 64)       { src = x1; cc = c; }
    else if (c < 128) { src = x2; cc = c - 64; }
    else              { src = x3; cc = c - 128; }
    out[tid] = src[((size_t)b * 64 + cc) * NPTS + n];
}

// h1216[b, c, n] = {broadcast g (1024), x1, x2, x3}
__global__ void build_h1216_kernel(const float* __restrict__ g, const float* __restrict__ x1,
                                   const float* __restrict__ x2, const float* __restrict__ x3,
                                   float* __restrict__ out)
{
    size_t tid = (size_t)blockIdx.x * 256 + threadIdx.x;
    if (tid >= (size_t)BATCH * 1216 * NPTS) return;
    int n = (int)(tid % NPTS); size_t r = tid / NPTS;
    int c = (int)(r % 1216);   int b = (int)(r / 1216);
    float v;
    if (c < 1024)      v = g[(size_t)b * 1024 + c];
    else if (c < 1088) v = x1[((size_t)b * 64 + (c - 1024)) * NPTS + n];
    else if (c < 1152) v = x2[((size_t)b * 64 + (c - 1088)) * NPTS + n];
    else               v = x3[((size_t)b * 64 + (c - 1152)) * NPTS + n];
    out[tid] = v;
}

// final 2-wide conv + transpose to (B, N, 2)
__global__ void final_kernel(const float* __restrict__ W9, const float* __restrict__ h,
                             float* __restrict__ out)
{
    int tid = blockIdx.x * 256 + threadIdx.x;   // B*N threads
    int n = tid % NPTS, b = tid / NPTS;
    const float* hb = h + (size_t)b * 256 * NPTS;
    float a0 = 0.f, a1 = 0.f;
    for (int c = 0; c < 256; ++c) {
        float v = hb[(size_t)c * NPTS + n];
        a0 += W9[c] * v;
        a1 += W9[256 + c] * v;
    }
    out[(size_t)tid * 2 + 0] = a0;
    out[(size_t)tid * 2 + 1] = a1;
}

// ---------------------------------------------------------------------------
extern "C" void kernel_launch(void* const* d_in, const int* in_sizes, int n_in,
                              void* d_out, int out_size, void* d_ws, size_t ws_size,
                              hipStream_t stream)
{
    const float* x  = (const float*)d_in[0];  // (4, 4096, 4)
    const float* w1 = (const float*)d_in[1];  // (64, 8)
    const float* w2 = (const float*)d_in[2];  // (64, 64)
    const float* w3 = (const float*)d_in[3];  // (64, 128)
    const float* w4 = (const float*)d_in[4];  // (64, 64)
    const float* w5 = (const float*)d_in[5];  // (64, 128)
    const float* w6 = (const float*)d_in[6];  // (1024, 192)
    const float* w7 = (const float*)d_in[7];  // (512, 1216)
    const float* w8 = (const float*)d_in[8];  // (256, 512)
    const float* w9 = (const float*)d_in[9];  // (2, 256)
    float* out = (float*)d_out;

    // --- workspace: two 84MB ping-pong buffers (fit in 192MB L2) + small ---
    char* ws = (char*)d_ws;
    size_t p = 0;
    auto alloc = [&](size_t bytes) { void* r = ws + p; p = (p + bytes + 255) & ~(size_t)255; return r; };
    int*   idx   = (int*)  alloc((size_t)BATCH * NPTS * KNN * sizeof(int));
    float* big1  = (float*)alloc((size_t)BATCH * 64 * NKCOL * sizeof(float)); // also h1216/h256/cat
    float* big2  = (float*)alloc((size_t)BATCH * 64 * NKCOL * sizeof(float)); // also g-pre-max/h512
    float* x1    = (float*)alloc((size_t)BATCH * 64 * NPTS * sizeof(float));
    float* x2    = (float*)alloc((size_t)BATCH * 64 * NPTS * sizeof(float));
    float* x3    = (float*)alloc((size_t)BATCH * 64 * NPTS * sizeof(float));
    float* tbias = (float*)alloc((size_t)BATCH * 64 * NPTS * sizeof(float));
    float* g     = (float*)alloc((size_t)BATCH * 1024 * sizeof(float));

    const dim3 knnGrid(NPTS / 256, BATCH);
    const int perPtBlocks = (BATCH * 64 * NPTS) / 256;                 // 4096
    const size_t gath64 = (size_t)BATCH * 64 * NKCOL;
    const size_t gath4  = (size_t)BATCH * 4 * NKCOL;
    const size_t e1Tot  = (size_t)BATCH * 64 * NKCOL;
    // x layout (B,N,4): batchStride=N*4, cStride=1, nStride=4
    // feature maps (B,64,N): batchStride=64*N, cStride=N, nStride=1

    // ---- stage 1 ----
    knn_kernel<<<knnGrid, 256, 0, stream>>>(x, idx, NPTS * 4, 1, 4);
    gather_kernel<<<(int)((gath4 + 255) / 256), 256, 0, stream>>>(x, idx, big2, 4, NPTS * 4, 1, 4, gath4);
    bias_t_kernel<<<perPtBlocks, 256, 0, stream>>>(w1, 8, 4, x, NPTS * 4, 1, 4, tbias);
    edge1_kernel<<<(int)((e1Tot + 255) / 256), 256, 0, stream>>>(w1, big2, tbias, big1);
    instnorm_lrelu_kernel<<<BATCH * 64, 256, 0, stream>>>(big1, NKCOL);
    gemm_wmma_kernel<0><<<dim3(NKCOL / 64, 1, BATCH), 256, 0, stream>>>(w2, 64, 64, 64, big1, NKCOL, big2, nullptr);
    instnorm_lrelu_kernel<<<BATCH * 64, 256, 0, stream>>>(big2, NKCOL);
    maxk_kernel<<<perPtBlocks, 256, 0, stream>>>(big2, x1, (size_t)BATCH * 64 * NPTS);

    // ---- stage 2 ----
    knn_kernel<<<knnGrid, 256, 0, stream>>>(x1, idx, 64 * NPTS, NPTS, 1);
    gather_kernel<<<(int)((gath64 + 255) / 256), 256, 0, stream>>>(x1, idx, big1, 64, 64 * NPTS, NPTS, 1, gath64);
    bias_t_kernel<<<perPtBlocks, 256, 0, stream>>>(w3, 128, 64, x1, 64 * NPTS, NPTS, 1, tbias);
    gemm_wmma_kernel<KNN><<<dim3(NKCOL / 64, 1, BATCH), 256, 0, stream>>>(w3, 128, 64, 64, big1, NKCOL, big2, tbias);
    instnorm_lrelu_kernel<<<BATCH * 64, 256, 0, stream>>>(big2, NKCOL);
    gemm_wmma_kernel<0><<<dim3(NKCOL / 64, 1, BATCH), 256, 0, stream>>>(w4, 64, 64, 64, big2, NKCOL, big1, nullptr);
    instnorm_lrelu_kernel<<<BATCH * 64, 256, 0, stream>>>(big1, NKCOL);
    maxk_kernel<<<perPtBlocks, 256, 0, stream>>>(big1, x2, (size_t)BATCH * 64 * NPTS);

    // ---- stage 3 ----
    knn_kernel<<<knnGrid, 256, 0, stream>>>(x2, idx, 64 * NPTS, NPTS, 1);
    gather_kernel<<<(int)((gath64 + 255) / 256), 256, 0, stream>>>(x2, idx, big1, 64, 64 * NPTS, NPTS, 1, gath64);
    bias_t_kernel<<<perPtBlocks, 256, 0, stream>>>(w5, 128, 64, x2, 64 * NPTS, NPTS, 1, tbias);
    gemm_wmma_kernel<KNN><<<dim3(NKCOL / 64, 1, BATCH), 256, 0, stream>>>(w5, 128, 64, 64, big1, NKCOL, big2, tbias);
    instnorm_lrelu_kernel<<<BATCH * 64, 256, 0, stream>>>(big2, NKCOL);
    maxk_kernel<<<perPtBlocks, 256, 0, stream>>>(big2, x3, (size_t)BATCH * 64 * NPTS);

    // ---- global feature ----
    const size_t catTot = (size_t)BATCH * 192 * NPTS;
    build_cat_kernel<<<(int)((catTot + 255) / 256), 256, 0, stream>>>(x1, x2, x3, big1);
    gemm_wmma_kernel<0><<<dim3(NPTS / 64, 1024 / 64, BATCH), 256, 0, stream>>>(w6, 192, 1024, 192, big1, NPTS, big2, nullptr);
    instnorm_lrelu_kernel<<<BATCH * 1024, 256, 0, stream>>>(big2, NPTS);
    maxn_kernel<<<BATCH * 1024, 256, 0, stream>>>(big2, g);

    // ---- segmentation head ----
    const size_t hTot = (size_t)BATCH * 1216 * NPTS;
    build_h1216_kernel<<<(int)((hTot + 255) / 256), 256, 0, stream>>>(g, x1, x2, x3, big1);
    gemm_wmma_kernel<0><<<dim3(NPTS / 64, 512 / 64, BATCH), 256, 0, stream>>>(w7, 1216, 512, 1216, big1, NPTS, big2, nullptr);
    instnorm_lrelu_kernel<<<BATCH * 512, 256, 0, stream>>>(big2, NPTS);
    gemm_wmma_kernel<0><<<dim3(NPTS / 64, 256 / 64, BATCH), 256, 0, stream>>>(w8, 512, 256, 512, big2, NPTS, big1, nullptr);
    instnorm_lrelu_kernel<<<BATCH * 256, 256, 0, stream>>>(big1, NPTS);
    final_kernel<<<(BATCH * NPTS) / 256, 256, 0, stream>>>(w9, big1, out);
}
